// UnifiedGNN_21887153341085
// MI455X (gfx1250) — matrix operand order repaired and checked by reading, not compiled, not run on hardware
//
#include <hip/hip_runtime.h>
#include <hip/hip_bf16.h>
#include <math.h>

typedef __attribute__((ext_vector_type(16))) _Float16 v16h;
typedef __attribute__((ext_vector_type(8)))  _Float16 v8h;
typedef __attribute__((ext_vector_type(8)))  float    v8f;

static constexpr int NP  = 12288;   // piece nodes
static constexpr int NSQ = 16384;   // square nodes
static constexpr int NB  = 256;     // graphs in batch
static constexpr int E_OCC_ = 12288, E_ATK_ = 49152, E_ADJ_ = 65536;

__device__ __forceinline__ float gelu_exact(float x) {
  return 0.5f * x * (1.0f + erff(x * 0.7071067811865476f));
}
// order-preserving float->uint map for atomic max on floats
__device__ __forceinline__ unsigned fkey(float f) {
  unsigned u = __float_as_uint(f);
  return (u & 0x80000000u) ? ~u : (u | 0x80000000u);
}
__device__ __forceinline__ float funkey(unsigned m) {
  return __uint_as_float((m & 0x80000000u) ? (m & 0x7FFFFFFFu) : ~m);
}

// ---------------- f32 -> f16 convert ----------------
__global__ void k_cvt16(const float* __restrict__ s, _Float16* __restrict__ d, int n) {
  int t = blockIdx.x * blockDim.x + threadIdx.x;
  if (t < n) d[t] = (_Float16)s[t];
}

// ---------------- input linear (tiny K) + GELU -> f16 ----------------
__global__ void k_lin(const float* __restrict__ X, const float* __restrict__ W,
                      const float* __restrict__ b, _Float16* __restrict__ Y,
                      int rows, int kin) {
  int t = blockIdx.x * blockDim.x + threadIdx.x;
  if (t >= rows * 128) return;
  int r = t >> 7, c = t & 127;
  float s = b[c];
  const float* xr = X + (size_t)r * kin;
  for (int k = 0; k < kin; ++k) s = fmaf(xr[k], W[k * 128 + c], s);
  Y[t] = (_Float16)gelu_exact(s);
}

// ---------------- WMMA GEMM: Y[M,N] = X[M,K](f16) @ W[K,N](f16) + bias, f32 out
// grid = (N/64, M/256), block = 256 (8 waves). Wave w -> rows [by*256+w*32, +32),
// cols [bx*64, +64). Two A fragments per k-step; each B fragment feeds 2 WMMAs.
// M % 256 == 0, N % 64 == 0, K % 32 == 0 guaranteed by caller; EXEC all-ones.
__global__ __launch_bounds__(256) void k_gemm(const _Float16* __restrict__ X,
    const _Float16* __restrict__ W, const float* __restrict__ bias,
    float* __restrict__ Y, int M, int N, int K) {
  const int lane  = threadIdx.x & 31;
  const int wave  = threadIdx.x >> 5;
  const int nBase = blockIdx.x * 64;
  const int mBase = blockIdx.y * 256 + wave * 32;
  const int kshift = (lane >> 4) << 3;          // A layout: hi lanes start at K=8 / K=24
  v8f a00 = {}, a01 = {}, a02 = {}, a03 = {};   // rows mBase..mBase+15
  v8f a10 = {}, a11 = {}, a12 = {}, a13 = {};   // rows mBase+16..mBase+31
  const _Float16* xrow0 = X + (size_t)(mBase + (lane & 15)) * K + kshift;
  const _Float16* xrow1 = xrow0 + (size_t)16 * K;
  for (int k0 = 0; k0 < K; k0 += 32) {
    // A fragments (16x32 f16): a[0..7]=X[m][k0+ks+0..7], a[8..15]=X[m][k0+ks+16..23]
    v8h lo0 = *reinterpret_cast<const v8h*>(xrow0 + k0);
    v8h hi0 = *reinterpret_cast<const v8h*>(xrow0 + k0 + 16);
    v8h lo1 = *reinterpret_cast<const v8h*>(xrow1 + k0);
    v8h hi1 = *reinterpret_cast<const v8h*>(xrow1 + k0 + 16);
    v16h fa0, fa1;
#pragma unroll
    for (int j = 0; j < 8; ++j) {
      fa0[j] = lo0[j]; fa0[8 + j] = hi0[j];
      fa1[j] = lo1[j]; fa1[8 + j] = hi1[j];
    }
    // B fragments (32x16 f16): lane holds column N=lane&15, K in same order as A (ks+0..7, ks+16..23)
    const _Float16* wcol = W + (size_t)k0 * N + nBase + (lane & 15);
    v16h b0, b1, b2, b3;
#pragma unroll
    for (int s = 0; s < 16; ++s) {
      const int kk = (s < 8) ? (kshift + s) : (16 + kshift + (s - 8));
      const _Float16* p = wcol + (size_t)kk * N;
      b0[s] = p[0]; b1[s] = p[16]; b2[s] = p[32]; b3[s] = p[48];
    }
    a00 = __builtin_amdgcn_wmma_f32_16x16x32_f16(false, fa0, false, b0, (short)0, a00, false, false);
    a10 = __builtin_amdgcn_wmma_f32_16x16x32_f16(false, fa1, false, b0, (short)0, a10, false, false);
    a01 = __builtin_amdgcn_wmma_f32_16x16x32_f16(false, fa0, false, b1, (short)0, a01, false, false);
    a11 = __builtin_amdgcn_wmma_f32_16x16x32_f16(false, fa1, false, b1, (short)0, a11, false, false);
    a02 = __builtin_amdgcn_wmma_f32_16x16x32_f16(false, fa0, false, b2, (short)0, a02, false, false);
    a12 = __builtin_amdgcn_wmma_f32_16x16x32_f16(false, fa1, false, b2, (short)0, a12, false, false);
    a03 = __builtin_amdgcn_wmma_f32_16x16x32_f16(false, fa0, false, b3, (short)0, a03, false, false);
    a13 = __builtin_amdgcn_wmma_f32_16x16x32_f16(false, fa1, false, b3, (short)0, a13, false, false);
  }
  // C/D layout: lane -> N=lane&15, VGPR i -> rows i (lanes<16) / i+8 (lanes>=16)
  const int nCol = lane & 15;
  const int rHi  = (lane >> 4) << 3;
  const float bv0 = bias[nBase + nCol],      bv1 = bias[nBase + 16 + nCol];
  const float bv2 = bias[nBase + 32 + nCol], bv3 = bias[nBase + 48 + nCol];
#pragma unroll
  for (int i = 0; i < 8; ++i) {
    float* y0 = Y + (size_t)(mBase + rHi + i) * N + nBase + nCol;
    y0[0]  = a00[i] + bv0;
    y0[16] = a01[i] + bv1;
    y0[32] = a02[i] + bv2;
    y0[48] = a03[i] + bv3;
    float* y1 = Y + (size_t)(mBase + 16 + rHi + i) * N + nBase + nCol;
    y1[0]  = a10[i] + bv0;
    y1[16] = a11[i] + bv1;
    y1[32] = a12[i] + bv2;
    y1[48] = a13[i] + bv3;
  }
}

// ---------------- GATv2 attention logits: att[h] . leaky_relu(xl[src]+xr[dst]) ----
__global__ void k_logits(const float* __restrict__ xl, const float* __restrict__ xr,
    const int* __restrict__ src, const int* __restrict__ dst,
    const float* __restrict__ att, float* __restrict__ logits,
    int E, int C, int HC) {
  int t = blockIdx.x * blockDim.x + threadIdx.x;
  if (t >= E * 4) return;
  int e = t >> 2, h = t & 3;
  const float* pl = xl + (size_t)src[e] * HC + h * C;
  const float* pr = xr + (size_t)dst[e] * HC + h * C;
  const float* pa = att + h * C;
  float s = 0.f;
  for (int c = 0; c < C; ++c) {
    float v = pl[c] + pr[c];
    v = (v > 0.f) ? v : 0.2f * v;
    s = fmaf(pa[c], v, s);
  }
  logits[t] = s;
}

__global__ void k_segmax(const float* __restrict__ logits, const int* __restrict__ dst,
                         unsigned* __restrict__ segmax, int E) {
  int t = blockIdx.x * blockDim.x + threadIdx.x;
  if (t >= E * 4) return;
  int e = t >> 2, h = t & 3;
  atomicMax(&segmax[dst[e] * 4 + h], fkey(logits[t]));
}

__global__ void k_expsum(float* __restrict__ logits, const int* __restrict__ dst,
                         const unsigned* __restrict__ segmax, float* __restrict__ segsum,
                         int E) {
  int t = blockIdx.x * blockDim.x + threadIdx.x;
  if (t >= E * 4) return;
  int e = t >> 2, h = t & 3;
  int d4 = dst[e] * 4 + h;
  float ex = expf(logits[t] - funkey(segmax[d4]));
  logits[t] = ex;
  atomicAdd(&segsum[d4], ex);
}

// conv1 scatter (concat heads, HC=512): out[dst] += alpha[h] * xl[src]
__global__ void k_scatter1(const float* __restrict__ xl, const float* __restrict__ ex,
    const float* __restrict__ segsum, const int* __restrict__ src,
    const int* __restrict__ dst, float* __restrict__ out, int E) {
  int t = blockIdx.x * blockDim.x + threadIdx.x;
  if (t >= E * 512) return;
  int e = t >> 9, j = t & 511, h = j >> 7;
  int s = src[e], d = dst[e];
  float alpha = ex[e * 4 + h] / (segsum[d * 4 + h] + 1e-16f);
  atomicAdd(&out[(size_t)d * 512 + j], alpha * xl[(size_t)s * 512 + j]);
}

// conv2 scatter (mean over heads, HC=1024, C=256)
__global__ void k_scatter2(const float* __restrict__ xl, const float* __restrict__ ex,
    const float* __restrict__ segsum, const int* __restrict__ src,
    const int* __restrict__ dst, float* __restrict__ out, int E) {
  int t = blockIdx.x * blockDim.x + threadIdx.x;
  if (t >= E * 256) return;
  int e = t >> 8, c = t & 255;
  int s = src[e], d = dst[e];
  const float* xs = xl + (size_t)s * 1024 + c;
  float acc = 0.f;
#pragma unroll
  for (int h = 0; h < 4; ++h) {
    float alpha = ex[e * 4 + h] / (segsum[d * 4 + h] + 1e-16f);
    acc = fmaf(alpha, xs[h * 256], acc);
  }
  atomicAdd(&out[(size_t)d * 256 + c], 0.25f * acc);
}

// conv1 epilogue: add the two dst-type biases, GELU, convert to f16 for conv2
__global__ void k_post1(const float* __restrict__ in, const float* __restrict__ b1,
                        const float* __restrict__ b2, _Float16* __restrict__ out, int rows) {
  int t = blockIdx.x * blockDim.x + threadIdx.x;
  if (t >= rows * 512) return;
  int j = t & 511;
  out[t] = (_Float16)gelu_exact(in[t] + b1[j] + b2[j]);
}

// global max pool over 48 contiguous pieces per graph; biases hoisted past max
__global__ void k_pool(const float* __restrict__ out2, const float* __restrict__ b1,
                       const float* __restrict__ b2, float* __restrict__ y) {
  int t = blockIdx.x * blockDim.x + threadIdx.x;
  if (t >= NB * 256) return;
  int g = t >> 8, c = t & 255;
  float m = -INFINITY;
  const float* p = out2 + (size_t)g * 48 * 256 + c;
  for (int i = 0; i < 48; ++i) m = fmaxf(m, p[i * 256]);
  y[t] = m + b1[c] + b2[c];
}

static inline dim3 g1(long n) { return dim3((unsigned)((n + 255) / 256)); }

extern "C" void kernel_launch(void* const* d_in, const int* in_sizes, int n_in,
                              void* d_out, int out_size, void* d_ws, size_t ws_size,
                              hipStream_t stream) {
  (void)in_sizes; (void)n_in; (void)out_size; (void)ws_size;
  // ---- inputs (setup_inputs insertion order; params flattened depth-first) ----
  const float* x_piece  = (const float*)d_in[0];
  const float* x_square = (const float*)d_in[1];
  const int* esrc[4] = { (const int*)d_in[2], (const int*)d_in[4], (const int*)d_in[6], (const int*)d_in[8] };
  const int* edst[4] = { (const int*)d_in[3], (const int*)d_in[5], (const int*)d_in[7], (const int*)d_in[9] };
  const int  eCnt[4] = { E_OCC_, E_OCC_, E_ATK_, E_ADJ_ };           // occ, rev, atk, adj
  const int  srcP[4] = { 1, 0, 1, 0 };  // src is piece?
  const int  dstP[4] = { 0, 1, 1, 0 };  // dst is piece?
  const float* lpW = (const float*)d_in[11]; const float* lpB = (const float*)d_in[12];
  const float* lsW = (const float*)d_in[13]; const float* lsB = (const float*)d_in[14];
  // gat params per type: {Wl=0, bl=1, Wr=2, br=3, att=4, bias=5}
  auto C1 = [&](int t, int f) { return (const float*)d_in[15 + 6 * t + f]; };
  auto C2 = [&](int t, int f) { return (const float*)d_in[39 + 6 * t + f]; };

  // ---- workspace carve ----
  char* ws = (char*)d_ws; size_t off = 0;
  auto carve = [&](size_t bytes) -> char* {
    char* p = ws + off; off = (off + bytes + 255) & ~(size_t)255; return p;
  };
  _Float16* h16p = (_Float16*)carve((size_t)NP  * 128 * 2);
  _Float16* h16s = (_Float16*)carve((size_t)NSQ * 128 * 2);
  _Float16 *w1l[4], *w1r[4];
  for (int t = 0; t < 4; ++t) {
    w1l[t] = (_Float16*)carve((size_t)128 * 512 * 2);
    w1r[t] = (_Float16*)carve((size_t)128 * 512 * 2);
  }
  _Float16 *w2l[2], *w2r[2];                    // [0]=rev, [1]=atk (occ/adj dead in conv2)
  for (int t = 0; t < 2; ++t) {
    w2l[t] = (_Float16*)carve((size_t)512 * 1024 * 2);
    w2r[t] = (_Float16*)carve((size_t)512 * 1024 * 2);
  }
  float*    xl     = (float*)carve((size_t)NSQ * 1024 * 4);   // max src proj
  float*    xr     = (float*)carve((size_t)NP  * 1024 * 4);   // max dst proj
  float*    logits = (float*)carve((size_t)E_ADJ_ * 4 * 4);
  unsigned* segmax = (unsigned*)carve((size_t)NSQ * 4 * 4);
  float*    segsum = (float*)carve((size_t)NSQ * 4 * 4);
  float*    out1p  = (float*)carve((size_t)NP  * 512 * 4);
  float*    out1s  = (float*)carve((size_t)NSQ * 512 * 4);
  _Float16* h1p    = (_Float16*)carve((size_t)NP  * 512 * 2);
  _Float16* h1s    = (_Float16*)carve((size_t)NSQ * 512 * 2);
  float*    out2   = (float*)carve((size_t)NP * 256 * 4);

  // ---- weight conversion to f16 ----
  for (int t = 0; t < 4; ++t) {
    k_cvt16<<<g1(128 * 512), 256, 0, stream>>>(C1(t, 0), w1l[t], 128 * 512);
    k_cvt16<<<g1(128 * 512), 256, 0, stream>>>(C1(t, 2), w1r[t], 128 * 512);
  }
  const int c2types[2] = { 1, 2 };              // rev, atk
  for (int i = 0; i < 2; ++i) {
    k_cvt16<<<g1(512 * 1024), 256, 0, stream>>>(C2(c2types[i], 0), w2l[i], 512 * 1024);
    k_cvt16<<<g1(512 * 1024), 256, 0, stream>>>(C2(c2types[i], 2), w2r[i], 512 * 1024);
  }

  // ---- input linears ----
  k_lin<<<g1((long)NP  * 128), 256, 0, stream>>>(x_piece,  lpW, lpB, h16p, NP, 12);
  k_lin<<<g1((long)NSQ * 128), 256, 0, stream>>>(x_square, lsW, lsB, h16s, NSQ, 13);

  // ---- conv1: four edge types, outputs summed per dst type ----
  hipMemsetAsync(out1p, 0, (size_t)NP  * 512 * 4, stream);
  hipMemsetAsync(out1s, 0, (size_t)NSQ * 512 * 4, stream);
  for (int t = 0; t < 4; ++t) {
    const _Float16* hs = srcP[t] ? h16p : h16s;  int Ms = srcP[t] ? NP : NSQ;
    const _Float16* hd = dstP[t] ? h16p : h16s;  int Md = dstP[t] ? NP : NSQ;
    k_gemm<<<dim3(512 / 64, Ms / 256), 256, 0, stream>>>(hs, w1l[t], C1(t, 1), xl, Ms, 512, 128);
    k_gemm<<<dim3(512 / 64, Md / 256), 256, 0, stream>>>(hd, w1r[t], C1(t, 3), xr, Md, 512, 128);
    int E = eCnt[t];
    k_logits<<<g1((long)E * 4), 256, 0, stream>>>(xl, xr, esrc[t], edst[t], C1(t, 4), logits, E, 128, 512);
    hipMemsetAsync(segmax, 0, (size_t)Md * 4 * 4, stream);
    hipMemsetAsync(segsum, 0, (size_t)Md * 4 * 4, stream);
    k_segmax<<<g1((long)E * 4), 256, 0, stream>>>(logits, edst[t], segmax, E);
    k_expsum<<<g1((long)E * 4), 256, 0, stream>>>(logits, edst[t], segmax, segsum, E);
    float* out = dstP[t] ? out1p : out1s;
    k_scatter1<<<g1((long)E * 512), 256, 0, stream>>>(xl, logits, segsum, esrc[t], edst[t], out, E);
  }
  // conv1 epilogue: piece gets rev+atk biases, square gets occ+adj biases; GELU; f16
  k_post1<<<g1((long)NP  * 512), 256, 0, stream>>>(out1p, C1(1, 5), C1(2, 5), h1p, NP);
  k_post1<<<g1((long)NSQ * 512), 256, 0, stream>>>(out1s, C1(0, 5), C1(3, 5), h1s, NSQ);

  // ---- conv2: only rev (square->piece) and atk (piece->piece); square dst is dead ----
  hipMemsetAsync(out2, 0, (size_t)NP * 256 * 4, stream);
  for (int i = 0; i < 2; ++i) {
    int t = c2types[i];
    const _Float16* hs = srcP[t] ? h1p : h1s;  int Ms = srcP[t] ? NP : NSQ;
    k_gemm<<<dim3(1024 / 64, Ms / 256), 256, 0, stream>>>(hs, w2l[i], C2(t, 1), xl, Ms, 1024, 512);
    k_gemm<<<dim3(1024 / 64, NP / 256), 256, 0, stream>>>(h1p, w2r[i], C2(t, 3), xr, NP, 1024, 512);
    int E = eCnt[t];
    k_logits<<<g1((long)E * 4), 256, 0, stream>>>(xl, xr, esrc[t], edst[t], C2(t, 4), logits, E, 256, 1024);
    hipMemsetAsync(segmax, 0, (size_t)NP * 4 * 4, stream);
    hipMemsetAsync(segsum, 0, (size_t)NP * 4 * 4, stream);
    k_segmax<<<g1((long)E * 4), 256, 0, stream>>>(logits, edst[t], segmax, E);
    k_expsum<<<g1((long)E * 4), 256, 0, stream>>>(logits, edst[t], segmax, segsum, E);
    k_scatter2<<<g1((long)E * 256), 256, 0, stream>>>(xl, logits, segsum, esrc[t], edst[t], out2, E);
  }

  // ---- global max pool over 48 contiguous pieces per graph ----
  k_pool<<<g1((long)NB * 256), 256, 0, stream>>>(out2, C2(1, 5), C2(2, 5), (float*)d_out);
}
